// Attention_18915035971740
// MI455X (gfx1250) — compile-verified, matrix operand order checked by reading, not run on hardware
//
#include <hip/hip_runtime.h>
#include <stdint.h>

// ---------------------------------------------------------------------------
// Fixed problem shapes (match reference): B=2, S=2048, H=2048, NH=32, NKV=8, HD=64
// ---------------------------------------------------------------------------
constexpr int Bb   = 2;
constexpr int Ss   = 2048;
constexpr int Hh   = 2048;
constexpr int NH   = 32;
constexpr int NKV  = 8;
constexpr int HD   = 64;
constexpr int MROWS = Bb * Ss;     // 4096 token rows
constexpr int QD    = NH * HD;     // 2048
constexpr int KD    = NKV * HD;    // 512

typedef __attribute__((ext_vector_type(16))) __bf16 v16bf;
typedef __attribute__((ext_vector_type(8)))  float  v8f;

union BF16x16 { v16bf v; uint4 u[2]; };
union F8      { v8f  v; float f[8]; };

// ---------------------------------------------------------------------------
// Workspace layout (bytes)
// ---------------------------------------------------------------------------
constexpr size_t OFF_XB = 0;                                        // X bf16 [4096,2048]
constexpr size_t OFF_WQ = OFF_XB + (size_t)MROWS * Hh * 2;          // Wq bf16 [2048,2048]
constexpr size_t OFF_WK = OFF_WQ + (size_t)QD * Hh * 2;             // Wk bf16 [512,2048]
constexpr size_t OFF_WV = OFF_WK + (size_t)KD * Hh * 2;             // Wv bf16 [512,2048]
constexpr size_t OFF_WO = OFF_WV + (size_t)KD * Hh * 2;             // Wo bf16 [2048,2048]
constexpr size_t OFF_QB = OFF_WO + (size_t)Hh * QD * 2;             // Q  bf16 [4096,2048]
constexpr size_t OFF_KB = OFF_QB + (size_t)MROWS * QD * 2;          // K  bf16 [4096,512]
constexpr size_t OFF_VT = OFF_KB + (size_t)MROWS * KD * 2;          // V^T bf16 [2*8*64, 2048]
constexpr size_t OFF_AB = OFF_VT + (size_t)Bb * NKV * HD * Ss * 2;  // attn out bf16 [4096,2048]

// ---------------------------------------------------------------------------
// WMMA helpers
// ---------------------------------------------------------------------------
__device__ __forceinline__ v8f wmma_bf16(v16bf a, v16bf b, v8f c) {
  return __builtin_amdgcn_wmma_f32_16x16x32_bf16(false, a, false, b, (short)0, c,
                                                 false, false);
}

// Load one 16x32 bf16 operand in the CDNA5 A/B VGPR layout.
// lane 0-15  : row (lane&15), K {0..7} then {16..23}
// lane 16-31 : row (lane&15), K {8..15} then {24..31}
// `base` points at (row0, k0); `ldk` is the row stride in elements.
__device__ __forceinline__ v16bf load_ak(const __bf16* base, int ldk, int lane) {
  int half = lane >> 4;
  int r    = lane & 15;
  const __bf16* p = base + (size_t)r * ldk + half * 8;
  BF16x16 t;
  t.u[0] = *(const uint4*)(p);        // K half*8 .. half*8+7
  t.u[1] = *(const uint4*)(p + 16);   // K 16+half*8 ..
  return t.v;
}

__device__ __forceinline__ float rowmax16(float v) {
  #pragma unroll
  for (int m = 1; m < 16; m <<= 1) v = fmaxf(v, __shfl_xor(v, m, 32));
  return v;
}
__device__ __forceinline__ float rowsum16(float v) {
  #pragma unroll
  for (int m = 1; m < 16; m <<= 1) v += __shfl_xor(v, m, 32);
  return v;
}

// ---------------------------------------------------------------------------
// fp32 -> bf16 cast
// ---------------------------------------------------------------------------
__global__ void cast_f32_bf16(const float* __restrict__ src,
                              __bf16* __restrict__ dst, int n) {
  for (int i = blockIdx.x * blockDim.x + threadIdx.x; i < n;
       i += gridDim.x * blockDim.x)
    dst[i] = (__bf16)src[i];
}

// ---------------------------------------------------------------------------
// WMMA GEMM:  C[m,n] = sum_k A[m,k] * W[n,k] + bias[n]
// A bf16 row-major [M,K], W bf16 row-major [N,K].
// MODE 0: bf16 out, row-major [M,N]
// MODE 1: bf16 out, V^T layout: out[(b*NKV*HD + n)*S + s],  m = b*S + s
// MODE 2: f32 out, row-major [M,N]
// Block: 256 thr = 8 waves (2 M x 4 N); wave tile 64x64 (16 WMMA / K-step);
// block tile 128 x 256; K-step 32; prefetch 4 K-steps ahead.
// ---------------------------------------------------------------------------
template <int MODE>
__global__ __launch_bounds__(256) void gemm_wmma(const __bf16* __restrict__ A,
                                                 const __bf16* __restrict__ W,
                                                 const float* __restrict__ bias,
                                                 void* __restrict__ out,
                                                 int M, int N, int K) {
  const int lane = threadIdx.x & 31;
  const int wid  = threadIdx.x >> 5;
  const int wm   = wid & 1;
  const int wn   = wid >> 1;
  const int row0 = blockIdx.y * 128 + wm * 64;
  const int col0 = blockIdx.x * 256 + wn * 64;
  const int r16  = lane & 15;

  F8 acc[4][4];
  #pragma unroll
  for (int t = 0; t < 4; ++t)
    #pragma unroll
    for (int u = 0; u < 4; ++u)
      #pragma unroll
      for (int r = 0; r < 8; ++r) acc[t][u].f[r] = 0.0f;

  #pragma unroll 2
  for (int k = 0; k < K; k += 32) {
    // prefetch 4 K-steps ahead (uniform branch; emits global_prefetch_b8)
    if (k + 128 < K) {
      #pragma unroll
      for (int t = 0; t < 4; ++t)
        __builtin_prefetch(A + (size_t)(row0 + t * 16 + r16) * K + k + 128, 0, 1);
      #pragma unroll
      for (int u = 0; u < 4; ++u)
        __builtin_prefetch(W + (size_t)(col0 + u * 16 + r16) * K + k + 128, 0, 1);
    }
    v16bf am[4], bm[4];
    #pragma unroll
    for (int t = 0; t < 4; ++t)
      am[t] = load_ak(A + (size_t)(row0 + t * 16) * K + k, K, lane);
    #pragma unroll
    for (int u = 0; u < 4; ++u)
      bm[u] = load_ak(W + (size_t)(col0 + u * 16) * K + k, K, lane);
    #pragma unroll
    for (int t = 0; t < 4; ++t)
      #pragma unroll
      for (int u = 0; u < 4; ++u)
        acc[t][u].v = wmma_bf16(am[t], bm[u], acc[t][u].v);
  }

  const int half = lane >> 4, ln = lane & 15;
  #pragma unroll
  for (int t = 0; t < 4; ++t) {
    #pragma unroll
    for (int u = 0; u < 4; ++u) {
      #pragma unroll
      for (int r = 0; r < 8; ++r) {
        int m = row0 + t * 16 + r + 8 * half;
        int n = col0 + u * 16 + ln;
        float v = acc[t][u].f[r] + (bias ? bias[n] : 0.0f);
        if (MODE == 0) {
          ((__bf16*)out)[(size_t)m * N + n] = (__bf16)v;
        } else if (MODE == 1) {
          int b = m / Ss, s = m - b * Ss;
          ((__bf16*)out)[((size_t)(b * (NKV * HD) + n)) * Ss + s] = (__bf16)v;
        } else {
          ((float*)out)[(size_t)m * N + n] = v;
        }
      }
    }
  }
}

// ---------------------------------------------------------------------------
// RoPE (matches reference exactly: out[d] = x[d]*cos[p, d%32] + rot[d]*sin[p, d%32],
//       rot[2i] = -x[2i+1], rot[2i+1] = x[2i]), applied in-place on bf16 buffer
// laid out [rows, dim] with per-head stride HD=64.
// ---------------------------------------------------------------------------
__global__ void rope_kernel(__bf16* __restrict__ x,
                            const float* __restrict__ cosT,
                            const float* __restrict__ sinT,
                            const int* __restrict__ positions,
                            int rows, int dim) {
  int pairsPerRow = dim >> 1;
  int total = rows * pairsPerRow;
  for (int idx = blockIdx.x * blockDim.x + threadIdx.x; idx < total;
       idx += gridDim.x * blockDim.x) {
    int m   = idx / pairsPerRow;
    int rem = idx - m * pairsPerRow;
    int hh  = rem >> 5;        // head
    int i   = rem & 31;        // pair within head
    int c0  = hh * 64 + 2 * i;
    int c1  = c0 + 1;
    int pos = positions[m];
    int i0  = (2 * i) & 31;
    int i1  = (2 * i + 1) & 31;
    float x0 = (float)x[(size_t)m * dim + c0];
    float x1 = (float)x[(size_t)m * dim + c1];
    float o0 = x0 * cosT[pos * 32 + i0] - x1 * sinT[pos * 32 + i0];
    float o1 = x1 * cosT[pos * 32 + i1] + x0 * sinT[pos * 32 + i1];
    x[(size_t)m * dim + c0] = (__bf16)o0;
    x[(size_t)m * dim + c1] = (__bf16)o1;
  }
}

// ---------------------------------------------------------------------------
// Flash-style causal GQA attention, one wave per (b, head, 16-row q tile).
// Q [4096, 2048] bf16, K [4096, 512] bf16, V^T [(b*8+hk)*64 + d][s] bf16.
// Online softmax in fp32; 32 keys per iteration (2 score sub-tiles).
// P is relaid C-layout -> A-layout through LDS for the P·V WMMAs.
// ---------------------------------------------------------------------------
__global__ __launch_bounds__(32) void attn_kernel(const __bf16* __restrict__ Qb,
                                                  const __bf16* __restrict__ Kb,
                                                  const __bf16* __restrict__ Vt,
                                                  __bf16* __restrict__ Ab) {
  const int qt   = blockIdx.x;       // 0..127 (16-row q tiles)
  const int h    = blockIdx.y;       // 0..31
  const int b    = blockIdx.z;       // 0..1
  const int lane = threadIdx.x;
  const int hk   = h / (NH / NKV);
  const int qbase = qt * 16;
  const int half = lane >> 4, ln = lane & 15;
  const float sc = 0.125f;           // 1/sqrt(HD)

  __shared__ __align__(16) __bf16 Pl[16 * 32];

  // Q tile operands (16 q rows x 64 hd), loaded once
  const __bf16* qptr = Qb + (size_t)(b * Ss + qbase) * QD + h * HD;
  v16bf qa0 = load_ak(qptr + 0,  QD, lane);   // hd 0..31
  v16bf qa1 = load_ak(qptr + 32, QD, lane);   // hd 32..63

  F8 acc[4];
  float mrow[8], lrow[8];
  #pragma unroll
  for (int t = 0; t < 4; ++t)
    #pragma unroll
    for (int r = 0; r < 8; ++r) acc[t].f[r] = 0.0f;
  #pragma unroll
  for (int r = 0; r < 8; ++r) { mrow[r] = -1e30f; lrow[r] = 0.0f; }

  const int ktiles = qt / 2 + 1;     // causal: only tiles with kbase <= qmax
  for (int kt = 0; kt < ktiles; ++kt) {
    const int kbase = kt * 32;
    const __bf16* kp = Kb + (size_t)(b * Ss + kbase) * KD + hk * HD;

    // scores: S0 = Q x K[0..15]^T, S1 = Q x K[16..31]^T (K dim = 64, 2 wmma each)
    F8 s0, s1;
    #pragma unroll
    for (int r = 0; r < 8; ++r) { s0.f[r] = 0.0f; s1.f[r] = 0.0f; }
    {
      v16bf kb00 = load_ak(kp, KD, lane);
      v16bf kb01 = load_ak(kp + 32, KD, lane);
      s0.v = wmma_bf16(qa0, kb00, s0.v);
      s0.v = wmma_bf16(qa1, kb01, s0.v);
      v16bf kb10 = load_ak(kp + (size_t)16 * KD, KD, lane);
      v16bf kb11 = load_ak(kp + (size_t)16 * KD + 32, KD, lane);
      s1.v = wmma_bf16(qa0, kb10, s1.v);
      s1.v = wmma_bf16(qa1, kb11, s1.v);
    }

    // causal mask + online softmax update
    F8 p0s, p1s;
    #pragma unroll
    for (int r = 0; r < 8; ++r) {
      int row  = qbase + r + 8 * half;
      int key0 = kbase + ln;
      int key1 = key0 + 16;
      float a0 = (key0 <= row) ? s0.f[r] * sc : -1e30f;
      float a1 = (key1 <= row) ? s1.f[r] * sc : -1e30f;
      float rm   = rowmax16(fmaxf(a0, a1));
      float mnew = fmaxf(mrow[r], rm);
      float alpha = __expf(mrow[r] - mnew);
      float p0 = __expf(a0 - mnew);
      float p1 = __expf(a1 - mnew);
      float rs = rowsum16(p0 + p1);
      lrow[r] = lrow[r] * alpha + rs;
      mrow[r] = mnew;
      #pragma unroll
      for (int t = 0; t < 4; ++t) acc[t].f[r] *= alpha;
      p0s.f[r] = p0;
      p1s.f[r] = p1;
    }

    // relay P (C-layout) -> LDS row-major [16 q][32 keys]
    __syncthreads();
    #pragma unroll
    for (int r = 0; r < 8; ++r) {
      Pl[(r + 8 * half) * 32 + ln]      = (__bf16)p0s.f[r];
      Pl[(r + 8 * half) * 32 + 16 + ln] = (__bf16)p1s.f[r];
    }
    __syncthreads();
    v16bf pa = load_ak(&Pl[0], 32, lane);  // A-layout 16q x 32keys

    // O += P x V  (V^T rows are d; contiguous in keys)
    const __bf16* vp = Vt + (size_t)((b * NKV + hk) * HD) * Ss + kbase;
    #pragma unroll
    for (int t = 0; t < 4; ++t) {
      v16bf vb = load_ak(vp + (size_t)(t * 16) * Ss, Ss, lane);
      acc[t].v = wmma_bf16(pa, vb, acc[t].v);
    }
  }

  // normalize + store bf16 to attn-out [b, s, h*64 + d]
  #pragma unroll
  for (int r = 0; r < 8; ++r) {
    float inv = 1.0f / lrow[r];
    int row = qbase + r + 8 * half;
    #pragma unroll
    for (int t = 0; t < 4; ++t) {
      float v = acc[t].f[r] * inv;
      Ab[(size_t)(b * Ss + row) * QD + h * HD + t * 16 + ln] = (__bf16)v;
    }
  }
}

// ---------------------------------------------------------------------------
// Host-side pipeline
// ---------------------------------------------------------------------------
extern "C" void kernel_launch(void* const* d_in, const int* in_sizes, int n_in,
                              void* d_out, int out_size, void* d_ws, size_t ws_size,
                              hipStream_t stream) {
  (void)in_sizes; (void)n_in; (void)out_size; (void)ws_size;

  const float* hs   = (const float*)d_in[0];
  const float* cosT = (const float*)d_in[1];
  const float* sinT = (const float*)d_in[2];
  const int*   pos  = (const int*)d_in[3];
  // d_in[4] = mask (causal; applied analytically in the kernel)
  const float* q_w  = (const float*)d_in[5];
  const float* q_b  = (const float*)d_in[6];
  const float* k_w  = (const float*)d_in[7];
  const float* k_b  = (const float*)d_in[8];
  const float* v_w  = (const float*)d_in[9];
  const float* v_b  = (const float*)d_in[10];
  const float* o_w  = (const float*)d_in[11];

  char* ws = (char*)d_ws;
  __bf16* Xb = (__bf16*)(ws + OFF_XB);
  __bf16* Wq = (__bf16*)(ws + OFF_WQ);
  __bf16* Wk = (__bf16*)(ws + OFF_WK);
  __bf16* Wv = (__bf16*)(ws + OFF_WV);
  __bf16* Wo = (__bf16*)(ws + OFF_WO);
  __bf16* Qb = (__bf16*)(ws + OFF_QB);
  __bf16* Kb = (__bf16*)(ws + OFF_KB);
  __bf16* Vt = (__bf16*)(ws + OFF_VT);
  __bf16* Ab = (__bf16*)(ws + OFF_AB);

  // 1) casts to bf16
  cast_f32_bf16<<<8192, 256, 0, stream>>>(hs,  Xb, MROWS * Hh);
  cast_f32_bf16<<<8192, 256, 0, stream>>>(q_w, Wq, QD * Hh);
  cast_f32_bf16<<<4096, 256, 0, stream>>>(k_w, Wk, KD * Hh);
  cast_f32_bf16<<<4096, 256, 0, stream>>>(v_w, Wv, KD * Hh);
  cast_f32_bf16<<<8192, 256, 0, stream>>>(o_w, Wo, Hh * QD);

  // 2) projections (WMMA GEMMs); block tile 128 x 256
  gemm_wmma<0><<<dim3(QD / 256, MROWS / 128), 256, 0, stream>>>(
      Xb, Wq, q_b, (void*)Qb, MROWS, QD, Hh);
  gemm_wmma<0><<<dim3(KD / 256, MROWS / 128), 256, 0, stream>>>(
      Xb, Wk, k_b, (void*)Kb, MROWS, KD, Hh);
  gemm_wmma<1><<<dim3(KD / 256, MROWS / 128), 256, 0, stream>>>(
      Xb, Wv, v_b, (void*)Vt, MROWS, KD, Hh);

  // 3) RoPE on Q and K (in-place, rewritten fresh every call)
  rope_kernel<<<16384, 256, 0, stream>>>(Qb, cosT, sinT, pos, MROWS, QD);
  rope_kernel<<<4096, 256, 0, stream>>>(Kb, cosT, sinT, pos, MROWS, KD);

  // 4) causal flash attention (1 wave / 16-row q tile / head)
  attn_kernel<<<dim3(Ss / 16, NH, Bb), 32, 0, stream>>>(Qb, Kb, Vt, Ab);

  // 5) output projection, fp32 result straight to d_out
  gemm_wmma<2><<<dim3(QD / 256, MROWS / 128), 256, 0, stream>>>(
      Ab, Wo, nullptr, d_out, MROWS, QD, Hh);
}
